// Decoder_21784074125982
// MI455X (gfx1250) — matrix operand order repaired and checked by reading, not compiled, not run on hardware
//
#include <hip/hip_runtime.h>
#include <math.h>

// ---------------- problem constants ----------------
#define BB_   16
#define T_    512
#define DM_   1024
#define H_    8
#define L_    6
#define D_    80
#define RF_   2
#define DK_   128
#define DIN_  4096
#define BT_   (BB_ * T_)      // 8192

// ---------------- WMMA types ----------------
typedef __attribute__((ext_vector_type(16))) __bf16 v16bf;
typedef __attribute__((ext_vector_type(8)))  float  v8f;
typedef unsigned short ushort_t;

union BF16x16 { v16bf v; ushort_t s[16]; uint4 q[2]; };

__device__ __forceinline__ ushort_t f2bf(float f) {
    union { float f; unsigned u; } x; x.f = f;
    unsigned u = x.u + 0x7FFFu + ((x.u >> 16) & 1u);   // round-to-nearest-even
    return (ushort_t)(u >> 16);
}

// =====================================================================
// Weight conversion: f32 -> bf16, 8 elements/thread (n % 8 == 0).
// =====================================================================
__global__ __launch_bounds__(256) void k_cvt_bf16(
    const float* __restrict__ src, ushort_t* __restrict__ dst, int n)
{
    const int i = (blockIdx.x * 256 + threadIdx.x) * 8;
    if (i >= n) return;
    const float4 a = *(const float4*)(src + i);
    const float4 b = *(const float4*)(src + i + 4);
    union { uint4 q; ushort_t s[8]; } o;
    o.s[0] = f2bf(a.x); o.s[1] = f2bf(a.y); o.s[2] = f2bf(a.z); o.s[3] = f2bf(a.w);
    o.s[4] = f2bf(b.x); o.s[5] = f2bf(b.y); o.s[6] = f2bf(b.z); o.s[7] = f2bf(b.w);
    *(uint4*)(dst + i) = o.q;
}

__global__ __launch_bounds__(256) void k_zero_us(ushort_t* __restrict__ dst, int n)
{
    const int i = blockIdx.x * 256 + threadIdx.x;
    if (i < n) dst[i] = 0;
}

// =====================================================================
// GEMM helpers: fragment loads in contiguous k order (k = 16*half + s).
// All loads are UNCONDITIONAL (caller guarantees Nload-padded buffers),
// keeping the hot loop branch-free so waitcnt can issue partial waits.
// =====================================================================
__device__ __forceinline__ void load_a_frags(
    BF16x16 (&af)[4], const ushort_t* __restrict__ A,
    int rowBase, int l, int K, int kA)
{
#pragma unroll
    for (int mb = 0; mb < 4; mb++) {
        const ushort_t* ap = A + (size_t)(rowBase + mb * 16 + l) * K + kA;
        af[mb].q[0] = *(const uint4*)(ap);
        af[mb].q[1] = *(const uint4*)(ap + 8);
    }
}

__device__ __forceinline__ void load_b_frags(
    BF16x16 (&bf)[4], const ushort_t* __restrict__ W,
    int colBase, int l, int K, int kB)
{
#pragma unroll
    for (int nb = 0; nb < 4; nb++) {
        const int n = colBase + nb * 16 + l;
        const ushort_t* wp = W + (size_t)n * K + kB;
        bf[nb].q[0] = *(const uint4*)(wp);
        bf[nb].q[1] = *(const uint4*)(wp + 8);
    }
}

__device__ __forceinline__ void wmma_step(
    v8f (&acc)[4][4], BF16x16 (&af)[4], BF16x16 (&bf)[4])
{
#pragma unroll
    for (int mb = 0; mb < 4; mb++)
#pragma unroll
        for (int nb = 0; nb < 4; nb++)
            acc[mb][nb] = __builtin_amdgcn_wmma_f32_16x16x32_bf16(
                false, af[mb].v, false, bf[nb].v,
                (short)0, acc[mb][nb], false, false);
}

// =====================================================================
// GEMM: C = act(A[M,K](bf16) @ W[N,K](bf16)^T + bias), f32 accum (WMMA).
// Block = 128 threads = 4 waves (2x2), block tile 128x128, wave tile 64x64.
// Branch-free double-buffered K loop: loads of step k+1 overlap WMMAs of
// step k. W must be valid (zero-padded) up to the 128-aligned column bound.
// Output: f32 (Cf) and/or bf16 (Cb), either may be null; stores guarded by N.
// =====================================================================
__global__ __launch_bounds__(128) void k_gemm_bf16(
    const ushort_t* __restrict__ A, const ushort_t* __restrict__ W,
    const float* __restrict__ bias,
    float* __restrict__ Cf, ushort_t* __restrict__ Cb,
    int M, int N, int K, int doRelu)
{
    (void)M;
    const int wave = threadIdx.x >> 5;
    const int lane = threadIdx.x & 31;
    const int hl   = lane >> 4;
    const int l    = lane & 15;
    const int wm   = wave >> 1;
    const int wn   = wave & 1;
    const int rowBase = blockIdx.y * 128 + wm * 64;
    const int colBase = blockIdx.x * 128 + wn * 64;

    v8f acc[4][4];
#pragma unroll
    for (int i = 0; i < 4; i++)
#pragma unroll
        for (int j = 0; j < 4; j++)
#pragma unroll
            for (int e = 0; e < 8; e++) acc[i][j][e] = 0.0f;

    BF16x16 a0[4], b0[4], a1[4], b1[4];
    const int steps = K / 32;

    // prologue: step 0 into set 0
    load_a_frags(a0, A, rowBase, l, K, hl * 16);
    load_b_frags(b0, W, colBase, l, K, hl * 16);

    // branch-free ping-pong pair loop
    for (int s = 0; s < steps - 1; s += 2) {
        const int k1 = (s + 1) * 32;
        const int s2 = (s + 2 < steps) ? (s + 2) : (steps - 1);  // clamp (uniform select)
        const int k2 = s2 * 32;
        load_a_frags(a1, A, rowBase, l, K, k1 + hl * 16);
        load_b_frags(b1, W, colBase, l, K, k1 + hl * 16);
        wmma_step(acc, a0, b0);
        load_a_frags(a0, A, rowBase, l, K, k2 + hl * 16);
        load_b_frags(b0, W, colBase, l, K, k2 + hl * 16);
        wmma_step(acc, a1, b1);
    }
    if (steps & 1)      // odd tail: last step resident in set 0
        wmma_step(acc, a0, b0);

    // ---- epilogue: D element (m,n): lane = half*16 + n%16, vgpr r -> m = half*8+r
#pragma unroll
    for (int nb = 0; nb < 4; nb++) {
        const int n = colBase + nb * 16 + l;
        if (n >= N) continue;
        const float bv = bias ? bias[n] : 0.0f;
#pragma unroll
        for (int mb = 0; mb < 4; mb++) {
#pragma unroll
            for (int r = 0; r < 8; r++) {
                const int m = rowBase + mb * 16 + hl * 8 + r;
                float v = acc[mb][nb][r] + bv;
                if (doRelu) v = fmaxf(v, 0.0f);
                const size_t idx = (size_t)m * N + n;
                if (Cf) Cf[idx] = v;
                if (Cb) Cb[idx] = f2bf(v);
            }
        }
    }
}

// =====================================================================
// Fused flash attention (no sqrt(dk) scale, matching reference).
// grid = (T/32, H, B), block = 256. Thread t: query qi=t/8, d-slice (t%8)*16.
// Q/K/V are f32; output written as bf16 (feeds only the next GEMM).
// =====================================================================
__global__ __launch_bounds__(256) void k_attn(
    const float* __restrict__ Q, const float* __restrict__ Kp,
    const float* __restrict__ Vp, ushort_t* __restrict__ Ob,
    int qStride, int kStride, int vStride,
    int qOff0, int kOff0, int vOff0,
    const int* __restrict__ dlen, int causal)
{
    __shared__ float sK[32 * DK_];
    __shared__ float sV[32 * DK_];

    const int bq  = blockIdx.x;
    const int h   = blockIdx.y;
    const int bb  = blockIdx.z;
    const int tid = threadIdx.x;
    const int qi  = tid >> 3;
    const int ds  = (tid & 7) * 16;
    const int qg  = bq * 32 + qi;
    const size_t qrow = (size_t)bb * T_ + qg;

    const float* qp = Q + qrow * qStride + qOff0 + h * DK_ + ds;
    float qv[16];
#pragma unroll
    for (int e = 0; e < 16; e++) qv[e] = qp[e];

    float accv[16];
#pragma unroll
    for (int e = 0; e < 16; e++) accv[e] = 0.0f;
    float mcur = -1e30f, lsum = 0.0f;
    const int len   = dlen[bb];
    const int ktEnd = causal ? (bq + 1) : (T_ / 32);

    for (int kt = 0; kt < ktEnd; kt++) {
        const int kbase = kt * 32;
        __syncthreads();
        for (int i = tid * 4; i < 32 * DK_; i += 256 * 4) {
            const int r = i >> 7;
            const int c = i & 127;
            *(float4*)(sK + r * DK_ + c) =
                *(const float4*)(Kp + ((size_t)bb * T_ + kbase + r) * kStride + kOff0 + h * DK_ + c);
            *(float4*)(sV + r * DK_ + c) =
                *(const float4*)(Vp + ((size_t)bb * T_ + kbase + r) * vStride + vOff0 + h * DK_ + c);
        }
        __syncthreads();

        for (int c8 = 0; c8 < 4; c8++) {
            float sc[8];
#pragma unroll
            for (int kk = 0; kk < 8; kk++) {
                const int kl = c8 * 8 + kk;
                const float* kr = sK + kl * DK_ + ds;
                float s = 0.0f;
#pragma unroll
                for (int e = 0; e < 16; e++) s = fmaf(qv[e], kr[e], s);
                s += __shfl_xor(s, 1, 8);
                s += __shfl_xor(s, 2, 8);
                s += __shfl_xor(s, 4, 8);
                const int kg = kbase + kl;
                const bool ok = (2 * kg < len) && (!causal || kg <= qg);
                sc[kk] = ok ? s : -1e30f;
            }
            float tmax = sc[0];
#pragma unroll
            for (int kk = 1; kk < 8; kk++) tmax = fmaxf(tmax, sc[kk]);
            const float mnew  = fmaxf(mcur, tmax);
            const float scale = __expf(mcur - mnew);
            lsum *= scale;
#pragma unroll
            for (int e = 0; e < 16; e++) accv[e] *= scale;
#pragma unroll
            for (int kk = 0; kk < 8; kk++) {
                const float p = __expf(sc[kk] - mnew);
                lsum += p;
                const float* vr = sV + (c8 * 8 + kk) * DK_ + ds;
#pragma unroll
                for (int e = 0; e < 16; e++) accv[e] = fmaf(p, vr[e], accv[e]);
            }
            mcur = mnew;
        }
    }
    const float inv = 1.0f / lsum;
    union { uint4 q[2]; ushort_t s[16]; } ob;
#pragma unroll
    for (int e = 0; e < 16; e++) ob.s[e] = f2bf(accv[e] * inv);
    uint4* op = (uint4*)(Ob + qrow * DM_ + h * DK_ + ds);
    op[0] = ob.q[0];
    op[1] = ob.q[1];
}

// =====================================================================
// LayerNorm kernels (row = 1024), block 256, 4 elems/thread.
// Write f32 master (X) and bf16 shadow (Xb) for the next GEMM.
// =====================================================================
__device__ __forceinline__ float posenc_val(int t, int c) {
    const float depth = (float)(c & ~1);
    const float denom = __powf(10000.0f, depth * (1.0f / (float)DM_));
    float ph = (float)t / denom;
    if ((c & 1) == 0) ph += 1.5707963267948966f;
    return __sinf(ph);
}

__global__ __launch_bounds__(256) void k_ln0(
    const float* __restrict__ Hs, float* __restrict__ X, ushort_t* __restrict__ Xb,
    const float* __restrict__ g, const float* __restrict__ b)
{
    __shared__ float red[256];
    const int row = blockIdx.x, tid = threadIdx.x;
    const int t = row & (T_ - 1);
    float v[4]; float s = 0.0f;
#pragma unroll
    for (int i = 0; i < 4; i++) {
        const int c = tid + i * 256;
        v[i] = Hs[(size_t)row * DM_ + c] + posenc_val(t, c);
        s += v[i];
    }
    red[tid] = s; __syncthreads();
    for (int st = 128; st > 0; st >>= 1) { if (tid < st) red[tid] += red[tid + st]; __syncthreads(); }
    const float mean = red[0] * (1.0f / DM_); __syncthreads();
    float s2 = 0.0f;
#pragma unroll
    for (int i = 0; i < 4; i++) { const float d = v[i] - mean; s2 += d * d; }
    red[tid] = s2; __syncthreads();
    for (int st = 128; st > 0; st >>= 1) { if (tid < st) red[tid] += red[tid + st]; __syncthreads(); }
    const float inv = rsqrtf(red[0] * (1.0f / DM_) + 1e-6f);
#pragma unroll
    for (int i = 0; i < 4; i++) {
        const int c = tid + i * 256;
        const float o = (v[i] - mean) * inv * g[c] + b[c];
        X[(size_t)row * DM_ + c]  = o;
        Xb[(size_t)row * DM_ + c] = f2bf(o);
    }
}

__global__ __launch_bounds__(256) void k_ln_res(
    const float* __restrict__ R, float* __restrict__ X, ushort_t* __restrict__ Xb,
    const float* __restrict__ g, const float* __restrict__ b)
{
    __shared__ float red[256];
    const int row = blockIdx.x, tid = threadIdx.x;
    float v[4]; float s = 0.0f;
#pragma unroll
    for (int i = 0; i < 4; i++) {
        const int c = tid + i * 256;
        v[i] = R[(size_t)row * DM_ + c] + X[(size_t)row * DM_ + c];
        s += v[i];
    }
    red[tid] = s; __syncthreads();
    for (int st = 128; st > 0; st >>= 1) { if (tid < st) red[tid] += red[tid + st]; __syncthreads(); }
    const float mean = red[0] * (1.0f / DM_); __syncthreads();
    float s2 = 0.0f;
#pragma unroll
    for (int i = 0; i < 4; i++) { const float d = v[i] - mean; s2 += d * d; }
    red[tid] = s2; __syncthreads();
    for (int st = 128; st > 0; st >>= 1) { if (tid < st) red[tid] += red[tid + st]; __syncthreads(); }
    const float inv = rsqrtf(red[0] * (1.0f / DM_) + 1e-6f);
#pragma unroll
    for (int i = 0; i < 4; i++) {
        const int c = tid + i * 256;
        const float o = (v[i] - mean) * inv * g[c] + b[c];
        X[(size_t)row * DM_ + c]  = o;
        Xb[(size_t)row * DM_ + c] = f2bf(o);
    }
}

// =====================================================================
// Data movement kernels
// =====================================================================
__global__ void k_build_tgt(const float* __restrict__ tg, ushort_t* __restrict__ TGTb)
{
    const int idx = blockIdx.x * 256 + threadIdx.x;
    if (idx >= BT_ * (D_ * RF_)) return;
    const int col = idx % (D_ * RF_);
    const int row = idx / (D_ * RF_);
    const int t  = row % T_;
    const int bb = row / T_;
    const int r = col / D_;
    const int d = col % D_;
    const int tt = t * RF_ + r - RF_;
    const float v = (tt >= 0) ? tg[((size_t)bb * D_ + d) * (T_ * RF_) + tt] : 0.0f;
    TGTb[idx] = f2bf(v);
}

__global__ void k_transpose_enc(const float* __restrict__ enc, ushort_t* __restrict__ ENCTb)
{
    const int idx = blockIdx.x * 256 + threadIdx.x;
    if (idx >= BT_ * DM_) return;
    const int c   = idx % DM_;
    const int row = idx / DM_;
    const int t  = row % T_;
    const int bb = row / T_;
    ENCTb[idx] = f2bf(enc[((size_t)bb * DM_ + c) * T_ + t]);
}

__global__ void k_out_reorder(const float* __restrict__ OUTB, float* __restrict__ out)
{
    const int idx = blockIdx.x * 256 + threadIdx.x;
    if (idx >= BB_ * D_ * T_ * RF_) return;
    const int time = idx % (T_ * RF_);
    const int d    = (idx / (T_ * RF_)) % D_;
    const int bb   = idx / (T_ * RF_ * D_);
    const int t = time >> 1;
    const int r = time & 1;
    out[idx] = OUTB[((size_t)bb * T_ + t) * (D_ * RF_) + r * D_ + d];
}

// =====================================================================
extern "C" void kernel_launch(void* const* d_in, const int* in_sizes, int n_in,
                              void* d_out, int out_size, void* d_ws, size_t ws_size,
                              hipStream_t stream)
{
    (void)in_sizes; (void)n_in; (void)out_size; (void)ws_size;
    const float* enc    = (const float*)d_in[0];
    const float* target = (const float*)d_in[1];
    const int*   dlen   = (const int*)d_in[2];
    // d_in[3] = max_len (compile-time T_ used)
    const float* prenet_w1 = (const float*)d_in[4];
    const float* prenet_b1 = (const float*)d_in[5];
    const float* prenet_w2 = (const float*)d_in[6];
    const float* prenet_b2 = (const float*)d_in[7];
    const float* ln0_g     = (const float*)d_in[8];
    const float* ln0_b     = (const float*)d_in[9];
    const float* self_Wqkv = (const float*)d_in[10];
    const float* self_Wfc  = (const float*)d_in[11];
    const float* self_ln_g = (const float*)d_in[12];
    const float* self_ln_b = (const float*)d_in[13];
    const float* cross_Wq  = (const float*)d_in[14];
    const float* cross_Wkv = (const float*)d_in[15];
    const float* cross_Wfc = (const float*)d_in[16];
    const float* cross_ln_g= (const float*)d_in[17];
    const float* cross_ln_b= (const float*)d_in[18];
    const float* ffn_w1    = (const float*)d_in[19];
    const float* ffn_b1    = (const float*)d_in[20];
    const float* ffn_w2    = (const float*)d_in[21];
    const float* ffn_b2    = (const float*)d_in[22];
    const float* ffn_ln_g  = (const float*)d_in[23];
    const float* ffn_ln_b  = (const float*)d_in[24];
    const float* out_w     = (const float*)d_in[25];
    const float* out_b     = (const float*)d_in[26];

    // ---- workspace layout: f32 region, then bf16 region ----
    float* X    = (float*)d_ws;                         // 8192*1024
    float* R    = X   + (size_t)BT_ * DM_;              // 8192*1024
    float* TMP  = R   + (size_t)BT_ * DM_;              // 8192*3072 (qkv | q + kv)
    float* OUTB = TMP + (size_t)BT_ * 3 * DM_;          // 8192*160

    ushort_t* Xb    = (ushort_t*)(OUTB + (size_t)BT_ * (D_ * RF_));
    ushort_t* Ob    = Xb    + (size_t)BT_ * DM_;
    ushort_t* F1b   = Ob    + (size_t)BT_ * DM_;        // 8192*4096
    ushort_t* TGTb  = F1b   + (size_t)BT_ * DIN_;       // 8192*160
    ushort_t* PRE1b = TGTb  + (size_t)BT_ * (D_ * RF_); // 8192*256
    ushort_t* ENCTb = PRE1b + (size_t)BT_ * 256;        // 8192*1024

    // bf16 weights (converted each call; deterministic)
    ushort_t* wcur = ENCTb + (size_t)BT_ * DM_;
    auto cvt = [&](const float* src, size_t n) -> ushort_t* {
        ushort_t* dst = wcur; wcur += n;
        k_cvt_bf16<<<((int)(n / 8) + 255) / 256, 256, 0, stream>>>(src, dst, (int)n);
        return dst;
    };
    ushort_t* w1b   = cvt(prenet_w1, (size_t)256 * (D_ * RF_));
    ushort_t* w2b   = cvt(prenet_w2, (size_t)DM_ * 256);
    ushort_t* qkvb  = cvt(self_Wqkv, (size_t)L_ * 3 * DM_ * DM_);
    ushort_t* sfcb  = cvt(self_Wfc,  (size_t)L_ * DM_ * DM_);
    ushort_t* cqb   = cvt(cross_Wq,  (size_t)L_ * DM_ * DM_);
    ushort_t* ckvb  = cvt(cross_Wkv, (size_t)L_ * 2 * DM_ * DM_);
    ushort_t* cfcb  = cvt(cross_Wfc, (size_t)L_ * DM_ * DM_);
    ushort_t* fw1b  = cvt(ffn_w1,    (size_t)L_ * DIN_ * DM_);
    ushort_t* fw2b  = cvt(ffn_w2,    (size_t)L_ * DM_ * DIN_);
    // out_w: pad to 256 rows (zero-filled) so N=160 fragment loads are legal
    ushort_t* outwb = wcur; wcur += (size_t)256 * DM_;
    k_zero_us<<<(256 * DM_ + 255) / 256, 256, 0, stream>>>(outwb, 256 * DM_);
    k_cvt_bf16<<<((160 * DM_) / 8 + 255) / 256, 256, 0, stream>>>(out_w, outwb, 160 * DM_);

    auto gemm = [&](const ushort_t* A, const ushort_t* W, const float* bias,
                    float* Cf, ushort_t* Cb, int N, int K, int relu) {
        dim3 g((N + 127) / 128, BT_ / 128);
        k_gemm_bf16<<<g, 128, 0, stream>>>(A, W, bias, Cf, Cb, BT_, N, K, relu);
    };

    // ---- prenet + posenc + LN0 ----
    k_build_tgt<<<(BT_ * D_ * RF_ + 255) / 256, 256, 0, stream>>>(target, TGTb);
    k_transpose_enc<<<(BT_ * DM_ + 255) / 256, 256, 0, stream>>>(enc, ENCTb);
    gemm(TGTb,  w1b, prenet_b1, nullptr, PRE1b, 256, D_ * RF_, 1);
    gemm(PRE1b, w2b, prenet_b2, R, nullptr, DM_, 256, 1);
    k_ln0<<<BT_, 256, 0, stream>>>(R, X, Xb, ln0_g, ln0_b);

    // ---- decoder layers ----
    for (int i = 0; i < L_; i++) {
        const ushort_t* Wqkv = qkvb + (size_t)i * 3 * DM_ * DM_;
        const ushort_t* Wfc  = sfcb + (size_t)i * DM_ * DM_;
        const ushort_t* cWq  = cqb  + (size_t)i * DM_ * DM_;
        const ushort_t* cWkv = ckvb + (size_t)i * 2 * DM_ * DM_;
        const ushort_t* cWfc = cfcb + (size_t)i * DM_ * DM_;
        const ushort_t* fw1  = fw1b + (size_t)i * DIN_ * DM_;
        const ushort_t* fw2  = fw2b + (size_t)i * DM_ * DIN_;
        const float* fb1 = ffn_b1 + (size_t)i * DIN_;
        const float* fb2 = ffn_b2 + (size_t)i * DM_;

        // self attention
        gemm(Xb, Wqkv, nullptr, TMP, nullptr, 3 * DM_, DM_, 0);
        k_attn<<<dim3(T_ / 32, H_, BB_), 256, 0, stream>>>(
            TMP, TMP, TMP, Ob, 3 * DM_, 3 * DM_, 3 * DM_,
            0, DM_, 2 * DM_, dlen, 1);
        gemm(Ob, Wfc, nullptr, R, nullptr, DM_, DM_, 0);
        k_ln_res<<<BT_, 256, 0, stream>>>(R, X, Xb, self_ln_g + i * DM_, self_ln_b + i * DM_);

        // cross attention (Q in TMP[0 .. BT*1024), KV at TMP + BT*1024)
        float* Qf  = TMP;
        float* KVf = TMP + (size_t)BT_ * DM_;
        gemm(Xb,    cWq,  nullptr, Qf,  nullptr, DM_,     DM_, 0);
        gemm(ENCTb, cWkv, nullptr, KVf, nullptr, 2 * DM_, DM_, 0);
        k_attn<<<dim3(T_ / 32, H_, BB_), 256, 0, stream>>>(
            Qf, KVf, KVf, Ob, DM_, 2 * DM_, 2 * DM_,
            0, 0, DM_, dlen, 0);
        gemm(Ob, cWfc, nullptr, R, nullptr, DM_, DM_, 0);
        k_ln_res<<<BT_, 256, 0, stream>>>(R, X, Xb, cross_ln_g + i * DM_, cross_ln_b + i * DM_);

        // FFN
        gemm(Xb,  fw1, fb1, nullptr, F1b, DIN_, DM_,  1);
        gemm(F1b, fw2, fb2, R, nullptr, DM_, DIN_, 0);
        k_ln_res<<<BT_, 256, 0, stream>>>(R, X, Xb, ffn_ln_g + i * DM_, ffn_ln_b + i * DM_);
    }

    // ---- output projection + reorder ----
    gemm(Xb, outwb, out_b, OUTB, nullptr, D_ * RF_, DM_, 0);
    k_out_reorder<<<(BB_ * D_ * T_ * RF_ + 255) / 256, 256, 0, stream>>>(OUTB, (float*)d_out);
}
